// Memory_45372034514972
// MI455X (gfx1250) — compile-verified, matrix operand order checked by reading, not run on hardware
//
#include <hip/hip_runtime.h>

// ---------------------------------------------------------------------------
// MI455X (gfx1250) chunked "Memory" recurrence.
// Phase 1: one persistent 512-thread workgroup; optimizer state in the WGP's
//          320KB LDS; all GEMMs via v_wmma_f32_16x16x32_bf16.
// Phase 2: chunk-parallel output GEMMs across the whole chip.
// All WMMA operands are bf16 in layouts where each lane's fragment is two
// contiguous 16B runs -> ds_load_b128 feeds, no per-use f32->bf16 converts.
// ---------------------------------------------------------------------------

typedef __bf16 bf16_t;
typedef __attribute__((ext_vector_type(16))) __bf16 v16bf;
typedef __attribute__((ext_vector_type(8)))  __bf16 v8bf;
typedef __attribute__((ext_vector_type(8)))  float  v8f;
typedef __attribute__((ext_vector_type(4)))  float  v4f;
typedef __attribute__((ext_vector_type(4)))  unsigned int v4u;

#define LLEN   2048
#define DDIM   64
#define CCH    16
#define FFDIM  256     // 4*d
#define NCHUNK 128     // L / C
#define NSLAB  8
#define SLABROWS 128

#define ALPHA_C 0.9f
#define THETA_C 0.44012666865176569f   // 0.95^16
#define TA_C    0.855f                 // 0.95 * 0.9
#define NS_A    3.4445f
#define NS_B    (-4.775f)
#define NS_C    2.0315f
#define NS_EPS  1e-5f
#define DDC_C   (1.0f/65536.0f)        // 1 / (B*H*C*dk)

__device__ __forceinline__ v8f wmma_bf16(v16bf a, v16bf b, v8f c) {
  return __builtin_amdgcn_wmma_f32_16x16x32_bf16(false, a, false, b,
                                                 (short)0, c, false, false);
}

// Row-major bf16 fragment: lane row0+(lane&15), two contiguous 16B runs at
// K = kb..kb+7 and 16+kb..16+kb+7 (kb = 8*(lane>>4)). Serves A (row=M) and
// transposed-B (row=N) operands.
__device__ __forceinline__ v16bf loadFrag_bf16(const bf16_t* base, int ld,
                                               int row0, int k0) {
  int lane = threadIdx.x & 31;
  int row = row0 + (lane & 15), kb = (lane >> 4) ? 8 : 0;
  const bf16_t* p = base + row * ld + k0 + kb;
  v8bf lo = *(const v8bf*)(p);
  v8bf hi = *(const v8bf*)(p + 16);
  return __builtin_shufflevector(lo, hi, 0, 1, 2, 3, 4, 5, 6, 7,
                                 8, 9, 10, 11, 12, 13, 14, 15);
}

// Same geometry from an f32 source: 4 x b128 + pack-cvt (used once per datum).
__device__ __forceinline__ v16bf loadFrag_f32(const float* base, int ld,
                                              int row0, int k0) {
  int lane = threadIdx.x & 31;
  int row = row0 + (lane & 15), kb = (lane >> 4) ? 8 : 0;
  const float* p = base + row * ld + k0 + kb;
  v4f x0 = *(const v4f*)(p);
  v4f x1 = *(const v4f*)(p + 4);
  v4f x2 = *(const v4f*)(p + 16);
  v4f x3 = *(const v4f*)(p + 20);
  v16bf a;
#pragma unroll
  for (int e = 0; e < 4; ++e) {
    a[e]      = (bf16_t)x0[e];
    a[4 + e]  = (bf16_t)x1[e];
    a[8 + e]  = (bf16_t)x2[e];
    a[12 + e] = (bf16_t)x3[e];
  }
  return a;
}

// Strided (column-gather) B fragment from row-major f32 [K x N]; cold path,
// used once per chunk to hoist the M1 operand into registers.
__device__ __forceinline__ v16bf loadB_f32_strided(const float* base, int ld,
                                                   int k0, int n0) {
  int lane = threadIdx.x & 31;
  int n = n0 + (lane & 15), kb = (lane >> 4) ? 8 : 0;
  v16bf b;
#pragma unroll
  for (int j = 0; j < 8; ++j) {
    int k = k0 + ((j < 4) ? (kb + 2 * j) : (16 + kb + 2 * (j - 4)));
    b[2 * j]     = (bf16_t)base[(k) * ld + n];
    b[2 * j + 1] = (bf16_t)base[(k + 1) * ld + n];
  }
  return b;
}

// ---- Newton-Schulz(5) on X[64x256] f32 in LDS -----------------------------
// Xb = row-major bf16 mirror (A/B^T feeds, zero cvt); XT/Yt = transposed bf16
// mirrors; Gb = bf16 Gram (stored transposed => vector stores; G symmetric).
__device__ void ns5(float* X, bf16_t* Xb, bf16_t* XT, bf16_t* Yt, bf16_t* Gb,
                    float* red, int tid, int wave) {
  const int lane = threadIdx.x & 31, ln = lane & 15, hi = lane >> 4;
  if (tid == 0) red[0] = 0.f;
  __syncthreads();
  float s = 0.f;
#pragma unroll
  for (int e = 0; e < 32; ++e) { float v = X[tid * 32 + e]; s += v * v; }
  atomicAdd(&red[0], s);
  __syncthreads();
  float scale = 1.f / (sqrtf(red[0]) + NS_EPS);
#pragma unroll
  for (int e = 0; e < 32; ++e) {
    int idx = tid * 32 + e;
    float v = X[idx] * scale;
    X[idx] = v;
    Xb[idx] = (bf16_t)v;
    XT[(idx & 255) * DDIM + (idx >> 8)] = (bf16_t)v;
  }
  __syncthreads();

  for (int it = 0; it < 5; ++it) {
    // G = X @ X^T  (16 tiles, one per wave); both operands from Xb, no cvt.
    {
      int gi = wave >> 2, gj = wave & 3;
      v8f acc = {};
#pragma unroll
      for (int ks = 0; ks < 8; ++ks) {
        v16bf a = loadFrag_bf16(Xb, FFDIM, gi * 16, ks * 32);
        v16bf b = loadFrag_bf16(Xb, FFDIM, gj * 16, ks * 32);
        acc = wmma_bf16(a, b, acc);
      }
      v8bf gv;
#pragma unroll
      for (int r = 0; r < 8; ++r) gv[r] = (bf16_t)acc[r];
      // store transposed (contiguous): legal because G is symmetric
      *(v8bf*)(Gb + (gj * 16 + ln) * DDIM + gi * 16 + 8 * hi) = gv;
    }
    __syncthreads();
    // Y = G @ X : A from Gb, B[k][n] = XT[n][k]; store Y transposed
#pragma unroll
    for (int j = 0; j < 4; ++j) {
      int id = wave * 4 + j, yi = id >> 4, yj = id & 15;
      v8f acc = {};
#pragma unroll
      for (int ks = 0; ks < 2; ++ks) {
        v16bf a = loadFrag_bf16(Gb, DDIM, yi * 16, ks * 32);
        v16bf b = loadFrag_bf16(XT, DDIM, yj * 16, ks * 32);
        acc = wmma_bf16(a, b, acc);
      }
      v8bf yv;
#pragma unroll
      for (int r = 0; r < 8; ++r) yv[r] = (bf16_t)acc[r];
      *(v8bf*)(Yt + (yj * 16 + ln) * DDIM + yi * 16 + 8 * hi) = yv;
    }
    __syncthreads();
    // X = a*X + b*Y + c*(G @ Y); refresh Xb and XT for next iteration
#pragma unroll
    for (int j = 0; j < 4; ++j) {
      int id = wave * 4 + j, yi = id >> 4, yj = id & 15;
      v8f acc = {};
#pragma unroll
      for (int ks = 0; ks < 2; ++ks) {
        v16bf a = loadFrag_bf16(Gb, DDIM, yi * 16, ks * 32);
        v16bf b = loadFrag_bf16(Yt, DDIM, yj * 16, ks * 32);
        acc = wmma_bf16(a, b, acc);
      }
      int col = yj * 16 + ln;
      v8bf yv = *(const v8bf*)(Yt + col * DDIM + yi * 16 + 8 * hi);
      v8bf xt;
#pragma unroll
      for (int r = 0; r < 8; ++r) {
        int idx = (yi * 16 + r + 8 * hi) * FFDIM + col;
        float xv = NS_A * X[idx] + NS_B * (float)yv[r] + NS_C * acc[r];
        X[idx] = xv;
        Xb[idx] = (bf16_t)xv;
        xt[r] = (bf16_t)xv;
      }
      *(v8bf*)(XT + col * DDIM + yi * 16 + 8 * hi) = xt;
    }
    __syncthreads();
  }
}

// ---- Phase 1: sequential state scan, 1 workgroup ---------------------------
__global__ __launch_bounds__(512) void mem_state_kernel(
    const float* __restrict__ Kg, const float* __restrict__ M1i,
    const float* __restrict__ M2i, bf16_t* __restrict__ ws,
    float* __restrict__ M1s) {
  extern __shared__ char smem[];
  float*  S1   = (float*)(smem);                 //  64KB S1   [64x256] f32
  float*  S2T  = (float*)(smem + 65536);         //  64KB S2^T [64x256] f32
  bf16_t* M2T  = (bf16_t*)(smem + 131072);       //  32KB M2^T [64x256] bf16
  char*   rgn  = smem + 163840;                  // 104KB aliased scratch
  bf16_t* slabDlpT = (bf16_t*)rgn;               //  stream: w*dlp^T [256x128]
  bf16_t* kRow = (bf16_t*)(rgn + 65536);         //  stream: k slab  [128x64]
  bf16_t* kT   = (bf16_t*)(rgn + 81920);         //  stream: k^T     [64x128]
  bf16_t* XT   = (bf16_t*)rgn;                   //  NS: X^T [256x64]
  bf16_t* Yt   = (bf16_t*)(rgn + 32768);         //  NS: Y^T [256x64]
  bf16_t* Xb   = (bf16_t*)(rgn + 65536);         //  NS: X   [64x256] bf16
  bf16_t* Gb   = (bf16_t*)(rgn + 98304);         //  NS: Gram [64x64] bf16
  float*  cs   = (float*)(smem + 270336);        //  1KB
  float*  rs2  = (float*)(smem + 271360);        //  1KB
  float*  red  = (float*)(smem + 272384);

  const int tid = threadIdx.x, wave = tid >> 5, lane = tid & 31;
  const int ln = lane & 15, hi = lane >> 4;
  const int fc = wave;                           // wave owns one f column block

  // init: M1 master state (f32, in workspace), M2^T bf16, S zero
#pragma unroll
  for (int e = 0; e < 32; ++e) {
    int idx = tid * 32 + e;                      // M2 [256x64]: idx = f*64 + k
    M1s[idx] = M1i[idx];
    M2T[(idx & 63) * FFDIM + (idx >> 6)] = (bf16_t)M2i[idx];
    S1[idx] = 0.f; S2T[idx] = 0.f;
  }
  __syncthreads();

  for (int t = 0; t < NCHUNK; ++t) {
    if (tid < FFDIM) {                 // rs2[f] = sum_k M2[f,k]; clear cs
      float s = 0.f;
      for (int k = 0; k < DDIM; ++k) s += (float)M2T[k * FFDIM + tid];
      rs2[tid] = s;
      cs[tid] = 0.f;
    }
    __syncthreads();

    // hoist M1 B-fragments for this wave's column block (reused 64x/chunk)
    v16bf bM1_0 = loadB_f32_strided(M1s, FFDIM, 0, fc * 16);
    v16bf bM1_1 = loadB_f32_strided(M1s, FFDIM, 32, fc * 16);
    v8f dacc[4] = {{}, {}, {}, {}};
    float csacc = 0.f;
    const float dscale = DDC_C * rs2[fc * 16 + ln];

    for (int sl = 0; sl < NSLAB; ++sl) {
      // stage k slab once: coalesced global b128, one cvt pass, kT for free
      {
        int r = tid >> 2;                        // slab row 0..127
        int db = (tid & 3) * 16;                 // d quarter
        int g = sl * 8 + (r >> 4);
        const float* src = Kg + ((size_t)g * LLEN + (size_t)t * CCH + (r & 15)) * DDIM + db;
        if (t + 1 < NCHUNK)
          __builtin_prefetch(Kg + ((size_t)g * LLEN + (size_t)(t + 1) * CCH + (r & 15)) * DDIM + db, 0, 1);
        v4f x0 = *(const v4f*)(src);
        v4f x1 = *(const v4f*)(src + 4);
        v4f x2 = *(const v4f*)(src + 8);
        v4f x3 = *(const v4f*)(src + 12);
        v8bf p0, p1;
#pragma unroll
        for (int e = 0; e < 4; ++e) {
          p0[e] = (bf16_t)x0[e]; p0[4 + e] = (bf16_t)x1[e];
          p1[e] = (bf16_t)x2[e]; p1[4 + e] = (bf16_t)x3[e];
        }
        *(v8bf*)(kRow + r * DDIM + db) = p0;
        *(v8bf*)(kRow + r * DDIM + db + 8) = p1;
#pragma unroll
        for (int e = 0; e < 8; ++e) {            // k^T from staging registers
          kT[(db + e) * SLABROWS + r] = p0[e];
          kT[(db + 8 + e) * SLABROWS + r] = p1[e];
        }
      }
      __syncthreads();
      // pass A: l1 = sigmoid(k@M1); emit w*dlp^T (one 16B store per lane/tile)
#pragma unroll
      for (int rt = 0; rt < 8; ++rt) {
        v16bf a0 = loadFrag_bf16(kRow, DDIM, rt * 16, 0);
        v16bf a1 = loadFrag_bf16(kRow, DDIM, rt * 16, 32);
        v8f c = {};
        c = wmma_bf16(a0, bM1_0, c);
        c = wmma_bf16(a1, bM1_1, c);
        v8bf dv;
#pragma unroll
        for (int r = 0; r < 8; ++r) {
          float wi = (float)(r + 8 * hi + 1);            // w[i] = i+1
          float l1 = 1.f / (1.f + __expf(-c[r]));
          csacc += wi * l1;
          dv[r] = (bf16_t)(wi * dscale * l1 * (1.f - l1));
        }
        *(v8bf*)(slabDlpT + (fc * 16 + ln) * SLABROWS + rt * 16 + 8 * hi) = dv;
      }
      __syncthreads();
      // pass B: dM1 += k^T @ (w*dlp): all operands contiguous b128
#pragma unroll
      for (int j = 0; j < 4; ++j) {
        int id = wave * 4 + j, dt = id >> 4, fc2 = id & 15;
#pragma unroll
        for (int ks = 0; ks < 4; ++ks) {
          v16bf a = loadFrag_bf16(kT, SLABROWS, dt * 16, ks * 32);
          v16bf b = loadFrag_bf16(slabDlpT, SLABROWS, fc2 * 16, ks * 32);
          dacc[j] = wmma_bf16(a, b, dacc[j]);
        }
      }
      __syncthreads();
    }

    atomicAdd(&cs[fc * 16 + ln], csacc);  // one atomic per lane per chunk
    // X1 = theta^C * S1 - theta*alpha * dM1  (element-exclusive, in place)
#pragma unroll
    for (int j = 0; j < 4; ++j) {
      int id = wave * 4 + j, dt = id >> 4, fc2 = id & 15;
#pragma unroll
      for (int r = 0; r < 8; ++r) {
        int idx = (dt * 16 + r + 8 * hi) * FFDIM + fc2 * 16 + ln;
        S1[idx] = THETA_C * S1[idx] - TA_C * dacc[j][r];
      }
    }
    __syncthreads();
    // X2^T = theta^C * S2^T - ta*ddc*cs[f]  (dM2 rows are identical)
#pragma unroll
    for (int e = 0; e < 32; ++e) {
      int idx = tid * 32 + e;
      S2T[idx] = THETA_C * S2T[idx] - TA_C * DDC_C * cs[idx & 255];
    }
    __syncthreads();

    ns5(S1, Xb, XT, Yt, Gb, red, tid, wave);    // S1  <- NS5(X1)
    ns5(S2T, Xb, XT, Yt, Gb, red, tid, wave);   // S2T <- NS5(X2^T)

    // M updates + bf16 export (M1 exported transposed for phase 2 B loads)
    bf16_t* wsM1T = ws + (size_t)t * 32768;     // [256x64]
    bf16_t* wsM2T = wsM1T + 16384;              // [64x256]
#pragma unroll
    for (int e = 0; e < 32; ++e) {
      int idx = tid * 32 + e;
      float m1 = ALPHA_C * M1s[idx] + S1[idx];
      M1s[idx] = m1;
      wsM1T[(idx & 255) * DDIM + (idx >> 8)] = (bf16_t)m1;
      float m2 = ALPHA_C * (float)M2T[idx] + S2T[idx];
      M2T[idx] = (bf16_t)m2;
      wsM2T[idx] = (bf16_t)m2;
    }
    __syncthreads();
  }
}

// ---- Phase 2: out = sigmoid(q @ M1n) @ M2n, chunk-parallel -----------------
__global__ __launch_bounds__(256) void mem_out_kernel(
    const float* __restrict__ Qg, const bf16_t* __restrict__ ws,
    float* __restrict__ out) {
  extern __shared__ char smem[];
  bf16_t* hbuf  = (bf16_t*)smem;              // 64KB h [128x256] bf16
  bf16_t* M1Tb  = (bf16_t*)(smem + 65536);    // 32KB [256x64]
  bf16_t* M2Tb  = (bf16_t*)(smem + 98304);    // 32KB [64x256]

  const int tid = threadIdx.x, wave = tid >> 5, lane = tid & 31;
  const int ln = lane & 15, hi = lane >> 4;
  const int t = blockIdx.x >> 3, sl = blockIdx.x & 7;
  const v4u* src1 = (const v4u*)(ws + (size_t)t * 32768);
  const v4u* src2 = src1 + 2048;

#pragma unroll
  for (int e = 0; e < 8; ++e) {               // b128 staging of both matrices
    ((v4u*)M1Tb)[tid + e * 256] = src1[tid + e * 256];
    ((v4u*)M2Tb)[tid + e * 256] = src2[tid + e * 256];
  }
  __syncthreads();

  // h = sigmoid(q @ M1n): wave owns one 16-row tile; q fragments hoisted
  {
    int rt = wave, g = sl * 8 + rt;
    const float* qrow = Qg + ((size_t)g * LLEN + (size_t)t * CCH) * DDIM;
    v16bf a0 = loadFrag_f32(qrow, DDIM, 0, 0);
    v16bf a1 = loadFrag_f32(qrow, DDIM, 0, 32);
#pragma unroll
    for (int fcc = 0; fcc < 16; ++fcc) {
      v16bf b0 = loadFrag_bf16(M1Tb, DDIM, fcc * 16, 0);
      v16bf b1 = loadFrag_bf16(M1Tb, DDIM, fcc * 16, 32);
      v8f c = {};
      c = wmma_bf16(a0, b0, c);
      c = wmma_bf16(a1, b1, c);
#pragma unroll
      for (int r = 0; r < 8; ++r) {
        float hv = 1.f / (1.f + __expf(-c[r]));
        hbuf[(rt * 16 + r + 8 * hi) * FFDIM + fcc * 16 + ln] = (bf16_t)hv;
      }
    }
  }
  __syncthreads();

  // out = h @ M2n : A fragments shared across the 4 output tiles
  {
    int rt = wave, g = sl * 8 + rt;
    v8f acc[4] = {{}, {}, {}, {}};
#pragma unroll
    for (int ks = 0; ks < 8; ++ks) {
      v16bf a = loadFrag_bf16(hbuf, FFDIM, rt * 16, ks * 32);
#pragma unroll
      for (int kt = 0; kt < 4; ++kt) {
        v16bf b = loadFrag_bf16(M2Tb, FFDIM, kt * 16, ks * 32);
        acc[kt] = wmma_bf16(a, b, acc[kt]);
      }
    }
#pragma unroll
    for (int kt = 0; kt < 4; ++kt)
#pragma unroll
      for (int r = 0; r < 8; ++r)
        out[((size_t)g * LLEN + (size_t)t * CCH + r + 8 * hi) * DDIM + kt * 16 + ln] = acc[kt][r];
  }
}

extern "C" void kernel_launch(void* const* d_in, const int* in_sizes, int n_in,
                              void* d_out, int out_size, void* d_ws, size_t ws_size,
                              hipStream_t stream) {
  (void)in_sizes; (void)n_in; (void)out_size; (void)ws_size;
  const float* Q = (const float*)d_in[0];
  const float* K = (const float*)d_in[1];
  // d_in[2] = V: never read numerically by the reference (ddiff is constant)
  const float* M1 = (const float*)d_in[3];
  const float* M2 = (const float*)d_in[4];
  bf16_t* ws = (bf16_t*)d_ws;                      // 8MB exports
  float* M1s = (float*)((char*)d_ws + 8u * 1024u * 1024u);  // +64KB f32 state
  float* out = (float*)d_out;

  const size_t smemA = 272400;           // < 320KB WGP LDS
  const size_t smemB = 131072;

  mem_state_kernel<<<1, 512, smemA, stream>>>(K, M1, M2, ws, M1s);
  mem_out_kernel<<<dim3(NCHUNK * NSLAB), 256, smemB, stream>>>(Q, ws, out);
}
